// SwinTransformer3D_59072980189206
// MI455X (gfx1250) — matrix operand order repaired and checked by reading, not compiled, not run on hardware
//
#include <hip/hip_runtime.h>

typedef __bf16 bf16;
typedef bf16     v16bf __attribute__((ext_vector_type(16)));
typedef float    v8f   __attribute__((ext_vector_type(8)));
typedef unsigned v8u   __attribute__((ext_vector_type(8)));

#define DEV static __device__ __forceinline__

// ---------- helpers ----------

DEV unsigned short f2bf_bits(float f) {
  unsigned u = __builtin_bit_cast(unsigned, f);
  unsigned r = (u + 0x7FFFu + ((u >> 16) & 1u)) >> 16;   // RNE
  return (unsigned short)r;
}

DEV bf16 f2bf(float f) { return __builtin_bit_cast(bf16, f2bf_bits(f)); }

// A fragment: 16x32 bf16, row-major source (4B aligned, even ld), src at (m0,k0)
DEV v16bf load_a_frag(const bf16* src, int ld, int lane) {
  v8u a;
  int m = lane & 15, h = lane >> 4;
  const unsigned* p = (const unsigned*)(src + m * ld);
#pragma unroll
  for (int r = 0; r < 8; ++r) {
    int k = (h << 4) + (r << 1);
    a[r] = p[k >> 1];
  }
  return __builtin_bit_cast(v16bf, a);
}

// B fragment where source holds B transposed: B[k][n] = src[n*ld + k]
DEV v16bf load_bT_frag(const bf16* src, int ld, int lane) {
  v8u b;
  int n = lane & 15, h = lane >> 4;
  const unsigned* p = (const unsigned*)(src + n * ld);
#pragma unroll
  for (int r = 0; r < 8; ++r) b[r] = p[h * 8 + r];
  return __builtin_bit_cast(v16bf, b);
}

// B fragment from pair-interleaved packed weights:
// packed[kp*ldN + n] = {B[2kp][n], B[2kp+1][n]}, p already offset to (kp0, n0)
DEV v16bf load_bp_frag(const unsigned* p, int ldN, int lane) {
  v8u b;
  int n = lane & 15, h = lane >> 4;
#pragma unroll
  for (int r = 0; r < 8; ++r) b[r] = p[(h * 8 + r) * ldN + n];
  return __builtin_bit_cast(v16bf, b);
}

DEV v8f wmma_bf16(v16bf a, v16bf b, v8f c) {
  return __builtin_amdgcn_wmma_f32_16x16x32_bf16(false, a, false, b, (short)0, c,
                                                 false, false);
}

// windowed row (0..200703) -> original flattened token index
// shift SS=(1,3,3), WS=(2,7,7), grid 8x8x8 windows per sample
DEV int row_to_tok(int row) {
  int wi = row / 98, nn = row % 98;
  int b  = wi >> 9, l = wi & 511;
  int wt = l >> 6, wh = (l >> 3) & 7, ww = l & 7;
  int i3 = nn / 49, j3 = (nn / 7) % 7, k3 = nn % 7;
  int t  = (wt * 2 + i3 + 1) & 15;
  int hh = (wh * 7 + j3 + 3) % 56;
  int wc = (ww * 7 + k3 + 3) % 56;
  return ((b * 16 + t) * 56 + hh) * 56 + wc;
}

// ---------- kernels ----------

// fp32 [K][N] -> pair-interleaved bf16 dwords [K/2][N]
__global__ void cvt_pack_kernel(const float* __restrict__ src,
                                unsigned* __restrict__ dst, int N, int total) {
  int i = blockIdx.x * 256 + threadIdx.x;
  if (i < total) {
    int kp = i / N, n = i % N;
    unsigned lo = f2bf_bits(src[(2 * kp) * N + n]);
    unsigned hi = f2bf_bits(src[(2 * kp + 1) * N + n]);
    dst[i] = lo | (hi << 16);
  }
}

// LN1 + cyclic shift + window partition, fp32 -> bf16.  One block (128 thr) per row.
__global__ void ln1_partition_kernel(const float* __restrict__ x,
                                     const float* __restrict__ g1,
                                     const float* __restrict__ b1,
                                     bf16* __restrict__ xw) {
  __shared__ float red[128];
  int row = blockIdx.x;
  int c = threadIdx.x;
  int tok = row_to_tok(row);
  float v = x[(size_t)tok * 128 + c];
  red[c] = v;
  __syncthreads();
  for (int s = 64; s > 0; s >>= 1) {
    if (c < s) red[c] += red[c + s];
    __syncthreads();
  }
  float mu = red[0] * (1.0f / 128.0f);
  __syncthreads();
  red[c] = v * v;
  __syncthreads();
  for (int s = 64; s > 0; s >>= 1) {
    if (c < s) red[c] += red[c + s];
    __syncthreads();
  }
  float var = red[0] * (1.0f / 128.0f) - mu * mu;
  float rs = rsqrtf(var + 1e-5f);
  xw[(size_t)row * 128 + c] = f2bf((v - mu) * rs * g1[c] + b1[c]);
}

// One block (256 thr = 8 waves) per (window, head).
__global__ void attn_kernel(const bf16* __restrict__ xw,
                            const unsigned* __restrict__ wqkv_p,  // [64][384]
                            const float* __restrict__ bqkv,
                            const float* __restrict__ rel_embed,
                            const int* __restrict__ rel_idx,
                            const float* __restrict__ mask,
                            bf16* __restrict__ attn_out) {
  __shared__ bf16 a_s[112 * 128] __attribute__((aligned(16)));  // xw, later P
  __shared__ bf16 q_s[112 * 32]  __attribute__((aligned(16)));
  __shared__ bf16 k_s[112 * 32]  __attribute__((aligned(16)));
  __shared__ bf16 v_s[32 * 128]  __attribute__((aligned(16)));  // V^T [d][row]

  int blk = blockIdx.x;
  int wi = blk >> 2;        // window 0..2047
  int head = blk & 3;
  int l = wi & 511;         // mask index
  int tid = threadIdx.x, lane = tid & 31, wid = tid >> 5;

  const bf16* srcw = xw + (size_t)wi * 98 * 128;
  for (int i = tid; i < 112 * 128; i += 256)
    a_s[i] = ((i >> 7) < 98) ? srcw[i] : f2bf(0.0f);
  for (int i = tid; i < 32 * 128; i += 256) v_s[i] = f2bf(0.0f);
  __syncthreads();

  // ---- qkv GEMM: 7 m-tiles x 6 col-groups (q0 q1 k0 k1 v0 v1), K loop 4 ----
  for (int tile = wid; tile < 42; tile += 8) {
    int mt = tile / 6, g = tile % 6;
    int sel = g >> 1, nb = (g & 1) << 4;
    int colbase = sel * 128 + head * 32 + nb;
    v8f acc = {};
#pragma unroll
    for (int kk = 0; kk < 4; ++kk) {
      v16bf a = load_a_frag(a_s + mt * 16 * 128 + kk * 32, 128, lane);
      v16bf b = load_bp_frag(wqkv_p + kk * 16 * 384 + colbase, 384, lane);
      acc = wmma_bf16(a, b, acc);
    }
    int n = lane & 15, h = lane >> 4;
    int d = nb + n;
    int mb = mt * 16 + h * 8;
    float bias = bqkv[colbase + n];
    if (sel == 2) {                 // V: transposed store [d][row]
      bf16* dst = v_s + d * 128 + mb;
#pragma unroll
      for (int r = 0; r < 8; ++r) dst[r] = f2bf(acc[r] + bias);
    } else {                        // Q (scaled) or K: [row][d]
      float mul = (sel == 0) ? 0.17677669529663687f : 1.0f;
      bf16* dst = ((sel == 0) ? q_s : k_s) + mb * 32 + d;
#pragma unroll
      for (int r = 0; r < 8; ++r) dst[r * 32] = f2bf((acc[r] + bias) * mul);
    }
  }
  __syncthreads();

  // ---- S + softmax (waves 0..6 own 16-row strips) ----
  if (wid < 7) {
    v8f s[7];
    v16bf aq = load_a_frag(q_s + wid * 16 * 32, 32, lane);
#pragma unroll
    for (int t = 0; t < 7; ++t) {
      v8f c = {};
      v16bf b = load_bT_frag(k_s + t * 16 * 32, 32, lane);
      s[t] = wmma_bf16(aq, b, c);
    }
    int n = lane & 15, h = lane >> 4;
    const float* mrow = mask + (size_t)l * 9604;
#pragma unroll
    for (int t = 0; t < 7; ++t) {
      int col = t * 16 + n;
#pragma unroll
      for (int r = 0; r < 8; ++r) {
        int rw = wid * 16 + h * 8 + r;
        float sv = s[t][r];
        if (col < 98 && rw < 98) {
          int ri = rel_idx[rw * 98 + col];
          sv += rel_embed[head * 507 + ri] + mrow[rw * 98 + col];
        } else {
          sv = -1e30f;
        }
        s[t][r] = sv;
      }
    }
#pragma unroll
    for (int r = 0; r < 8; ++r) {
      float mx = -1e30f;
#pragma unroll
      for (int t = 0; t < 7; ++t) mx = fmaxf(mx, s[t][r]);
      for (int off = 1; off < 16; off <<= 1) mx = fmaxf(mx, __shfl_xor(mx, off, 32));
      float sum = 0.0f;
#pragma unroll
      for (int t = 0; t < 7; ++t) {
        float e = expf(s[t][r] - mx);
        s[t][r] = e;
        sum += e;
      }
      for (int off = 1; off < 16; off <<= 1) sum += __shfl_xor(sum, off, 32);
      float inv = 1.0f / sum;
#pragma unroll
      for (int t = 0; t < 7; ++t) s[t][r] *= inv;
    }
    // write P (bf16) into a_s [112][128]; zero pad cols 112..127
#pragma unroll
    for (int t = 0; t < 7; ++t) {
      int col = t * 16 + n;
      bf16* dst = a_s + (wid * 16 + h * 8) * 128 + col;
#pragma unroll
      for (int r = 0; r < 8; ++r) dst[r * 128] = f2bf(s[t][r]);
    }
    for (int i = lane; i < 256; i += 32) {
      int rw = wid * 16 + (i >> 4);
      a_s[rw * 128 + 112 + (i & 15)] = f2bf(0.0f);
    }
  }
  __syncthreads();

  // ---- out = P @ V : 7 m-tiles x 2 n-tiles, K loop 4 (K padded to 128) ----
  for (int tile = wid; tile < 14; tile += 8) {
    int mt = tile >> 1, nt = tile & 1;
    v8f acc = {};
#pragma unroll
    for (int kk = 0; kk < 4; ++kk) {
      v16bf a = load_a_frag(a_s + mt * 16 * 128 + kk * 32, 128, lane);
      v16bf b = load_bT_frag(v_s + (nt * 16) * 128 + kk * 32, 128, lane);
      acc = wmma_bf16(a, b, acc);
    }
    int n = lane & 15, h = lane >> 4;
    int c = head * 32 + nt * 16 + n;
    int mb = mt * 16 + h * 8;
    bf16* dst = attn_out + ((size_t)wi * 98 + mb) * 128 + c;
    if (mt < 6) {                    // all 16 rows valid
#pragma unroll
      for (int r = 0; r < 8; ++r) dst[(size_t)r * 128] = f2bf(acc[r]);
    } else {                         // rows 96..111: only 96,97 valid
#pragma unroll
      for (int r = 0; r < 8; ++r)
        if (mb + r < 98) dst[(size_t)r * 128] = f2bf(acc[r]);
    }
  }
}

// proj GEMM (64-row tiles) + scatter to original coords + residual + LN2
__global__ void proj_ln2_kernel(const bf16* __restrict__ attn_o,
                                const unsigned* __restrict__ wproj_p,  // [64][128]
                                const float* __restrict__ bproj,
                                const float* __restrict__ x,
                                const float* __restrict__ g2,
                                const float* __restrict__ b2,
                                float* __restrict__ xres,
                                bf16* __restrict__ xn2) {
  __shared__ bf16 a_s[64 * 128] __attribute__((aligned(16)));
  __shared__ float o_s[64 * 128];
  int tid = threadIdx.x, lane = tid & 31, wid = tid >> 5;
  int m0 = blockIdx.x * 64;

  for (int i = tid; i < 64 * 128; i += 256) a_s[i] = attn_o[(size_t)m0 * 128 + i];
  __syncthreads();

  for (int tile = wid; tile < 32; tile += 8) {
    int mt = tile >> 3, nt = tile & 7;
    v8f acc = {};
#pragma unroll
    for (int kk = 0; kk < 4; ++kk) {
      v16bf a = load_a_frag(a_s + mt * 16 * 128 + kk * 32, 128, lane);
      v16bf b = load_bp_frag(wproj_p + kk * 16 * 128 + nt * 16, 128, lane);
      acc = wmma_bf16(a, b, acc);
    }
    int n = lane & 15, h = lane >> 4;
    int col = nt * 16 + n;
    float bias = bproj[col];
    float* dst = o_s + (mt * 16 + h * 8) * 128 + col;
#pragma unroll
    for (int r = 0; r < 8; ++r) dst[r * 128] = acc[r] + bias;
  }
  __syncthreads();

  // residual + LN2: 4 threads per row
  int row = tid >> 2, qq = tid & 3;
  int tok = row_to_tok(m0 + row);
  float s1 = 0.0f, s2 = 0.0f;
  for (int c = qq * 32; c < qq * 32 + 32; ++c) {
    float v = o_s[row * 128 + c] + x[(size_t)tok * 128 + c];
    o_s[row * 128 + c] = v;
    s1 += v;
    s2 += v * v;
  }
  s1 += __shfl_xor(s1, 1, 32); s1 += __shfl_xor(s1, 2, 32);
  s2 += __shfl_xor(s2, 1, 32); s2 += __shfl_xor(s2, 2, 32);
  float mu = s1 * (1.0f / 128.0f);
  float var = s2 * (1.0f / 128.0f) - mu * mu;
  float rs = rsqrtf(var + 1e-5f);
  for (int c = qq * 32; c < qq * 32 + 32; ++c) {
    float v = o_s[row * 128 + c];
    xres[(size_t)tok * 128 + c] = v;
    xn2[(size_t)tok * 128 + c] = f2bf((v - mu) * rs * g2[c] + b2[c]);
  }
}

// fused MLP: 32-row tiles, hidden (32x512 bf16) in LDS, exact GELU, + residual
__global__ void mlp_kernel(const bf16* __restrict__ xn2,
                           const unsigned* __restrict__ wfc1_p,  // [64][512]
                           const float* __restrict__ bfc1,
                           const unsigned* __restrict__ wfc2_p,  // [256][128]
                           const float* __restrict__ bfc2,
                           const float* __restrict__ xres,
                           float* __restrict__ out) {
  __shared__ bf16 a_s[32 * 128] __attribute__((aligned(16)));
  __shared__ bf16 h_s[32 * 512] __attribute__((aligned(16)));
  int tid = threadIdx.x, lane = tid & 31, wid = tid >> 5;
  int m0 = blockIdx.x * 32;

  for (int i = tid; i < 32 * 128; i += 256) a_s[i] = xn2[(size_t)m0 * 128 + i];
  __syncthreads();

  // fc1 + GELU: 2 m-tiles x 32 n-tiles
  for (int tile = wid; tile < 64; tile += 8) {
    int mt = tile >> 5, nt = tile & 31;
    v8f acc = {};
#pragma unroll
    for (int kk = 0; kk < 4; ++kk) {
      v16bf a = load_a_frag(a_s + mt * 16 * 128 + kk * 32, 128, lane);
      v16bf b = load_bp_frag(wfc1_p + kk * 16 * 512 + nt * 16, 512, lane);
      acc = wmma_bf16(a, b, acc);
    }
    int n = lane & 15, h = lane >> 4;
    int col = nt * 16 + n;
    float bias = bfc1[col];
    bf16* dst = h_s + (mt * 16 + h * 8) * 512 + col;
#pragma unroll
    for (int r = 0; r < 8; ++r) {
      float v = acc[r] + bias;
      v = 0.5f * v * (1.0f + erff(v * 0.7071067811865475f));
      dst[r * 512] = f2bf(v);
    }
  }
  __syncthreads();

  // fc2 + residual: 2 m-tiles x 8 n-tiles, K = 512
  for (int tile = wid; tile < 16; tile += 8) {
    int mt = tile >> 3, nt = tile & 7;
    v8f acc = {};
#pragma unroll
    for (int kk = 0; kk < 16; ++kk) {
      v16bf a = load_a_frag(h_s + mt * 16 * 512 + kk * 32, 512, lane);
      v16bf b = load_bp_frag(wfc2_p + kk * 16 * 128 + nt * 16, 128, lane);
      acc = wmma_bf16(a, b, acc);
    }
    int n = lane & 15, h = lane >> 4;
    int col = nt * 16 + n;
    float bias = bfc2[col];
    size_t gbase = (size_t)(m0 + mt * 16 + h * 8) * 128 + col;
#pragma unroll
    for (int r = 0; r < 8; ++r)
      out[gbase + (size_t)r * 128] = acc[r] + bias + xres[gbase + (size_t)r * 128];
  }
}

// ---------- launch ----------

extern "C" void kernel_launch(void* const* d_in, const int* in_sizes, int n_in,
                              void* d_out, int out_size, void* d_ws, size_t ws_size,
                              hipStream_t stream) {
  (void)in_sizes; (void)n_in; (void)out_size; (void)ws_size;

  const float* x         = (const float*)d_in[0];
  const float* g1        = (const float*)d_in[1];
  const float* b1        = (const float*)d_in[2];
  const float* w_qkv     = (const float*)d_in[3];
  const float* b_qkv     = (const float*)d_in[4];
  const float* rel_embed = (const float*)d_in[5];
  const float* w_proj    = (const float*)d_in[6];
  const float* b_proj    = (const float*)d_in[7];
  const float* g2        = (const float*)d_in[8];
  const float* b2        = (const float*)d_in[9];
  const float* w_fc1     = (const float*)d_in[10];
  const float* b_fc1     = (const float*)d_in[11];
  const float* w_fc2     = (const float*)d_in[12];
  const float* b_fc2     = (const float*)d_in[13];
  const float* attn_mask = (const float*)d_in[14];
  const int*   rel_idx   = (const int*)d_in[15];
  float* out = (float*)d_out;

  const size_t ROWS = 200704;  // B*T*H*W tokens == Bw*98 windowed rows
  char* ws = (char*)d_ws;
  bf16* xw     = (bf16*)ws;      ws += ROWS * 128 * 2;
  bf16* attn_o = (bf16*)ws;      ws += ROWS * 128 * 2;
  float* xres  = (float*)ws;     ws += ROWS * 128 * 4;
  bf16* xn2    = (bf16*)ws;      ws += ROWS * 128 * 2;
  unsigned* wqkv_p  = (unsigned*)ws; ws += 64 * 384 * 4;   // pair-packed weights
  unsigned* wproj_p = (unsigned*)ws; ws += 64 * 128 * 4;
  unsigned* wfc1_p  = (unsigned*)ws; ws += 64 * 512 * 4;
  unsigned* wfc2_p  = (unsigned*)ws; ws += 256 * 128 * 4;

  cvt_pack_kernel<<<(24576 + 255) / 256, 256, 0, stream>>>(w_qkv, wqkv_p, 384, 24576);
  cvt_pack_kernel<<<(8192  + 255) / 256, 256, 0, stream>>>(w_proj, wproj_p, 128, 8192);
  cvt_pack_kernel<<<(32768 + 255) / 256, 256, 0, stream>>>(w_fc1, wfc1_p, 512, 32768);
  cvt_pack_kernel<<<(32768 + 255) / 256, 256, 0, stream>>>(w_fc2, wfc2_p, 128, 32768);

  ln1_partition_kernel<<<200704, 128, 0, stream>>>(x, g1, b1, xw);
  attn_kernel<<<8192, 256, 0, stream>>>(xw, wqkv_p, b_qkv, rel_embed, rel_idx,
                                        attn_mask, attn_o);
  proj_ln2_kernel<<<3136, 256, 0, stream>>>(attn_o, wproj_p, b_proj, x, g2, b2,
                                            xres, xn2);
  mlp_kernel<<<6272, 256, 0, stream>>>(xn2, wfc1_p, b_fc1, wfc2_p, b_fc2, xres, out);
}
